// LAS_53644141527185
// MI455X (gfx1250) — compile-verified, matrix operand order checked by reading, not run on hardware
//
#include <hip/hip_runtime.h>
#include <hip/hip_bf16.h>
#include <stdint.h>

typedef __bf16 bf16_t;
typedef __attribute__((ext_vector_type(16))) __bf16 v16bf;
typedef __attribute__((ext_vector_type(8)))  __bf16 v8bf;
typedef __attribute__((ext_vector_type(8)))  float  v8f;

#define WMMA_BF16(A, B, C) \
  __builtin_amdgcn_wmma_f32_16x16x32_bf16(false, (A), false, (B), (short)0, (C), false, false)

__device__ __forceinline__ float sigf(float x) { return 1.f / (1.f + __expf(-x)); }

// ---------------------------------------------------------------------------
// CDNA5 16-bit A/B fragment loader (16x32 A, 32x16 B), per ISA 7.12.2:
// lane[3:0] = row (M for A, N for B), lane[4] = K-half selector.
// Each half is two contiguous 8x-bf16 (16B) runs -> ds_load_b128 / global_load_b128.
// ---------------------------------------------------------------------------
__device__ __forceinline__ v16bf load_frag16(const bf16_t* base, int row0, int stride, int kbase) {
  int lane  = threadIdx.x & 31;
  int r     = lane & 15;
  int khalf = lane >> 4;
  const bf16_t* p = base + (size_t)(row0 + r) * stride + kbase + khalf * 8;
  v16bf f;
#pragma unroll
  for (int e = 0; e < 8; ++e) f[e] = p[e];
  p += 16;
#pragma unroll
  for (int e = 0; e < 8; ++e) f[8 + e] = p[e];
  return f;
}

// ---------------------------------------------------------------------------
// Generic WMMA GEMM:  Out[(t,b), n] = sum_k Xrow(t,b)[k] * W[n*ldw + k] + bias1[n] (+bias2[n])
// 256 threads (8 waves), 64x64 tile, K staged as 64x32 bf16 LDS tiles.
// Uniform fast path for full K-chunks: float4 x2 loads -> 8x bf16 -> one 16B LDS store.
// ---------------------------------------------------------------------------
__global__ __launch_bounds__(256)
void k_gemm(const float* __restrict__ X, long xs_t, long xs_b,
            const float* __restrict__ W, int ldw,
            const float* __restrict__ bias1, const float* __restrict__ bias2,
            float* __restrict__ Out, int N, int K)
{
  __shared__ __align__(16) bf16_t xs[64 * 32];
  __shared__ __align__(16) bf16_t wt[64 * 32];
  int tid = threadIdx.x, lane = tid & 31, wid = tid >> 5;
  int m0 = blockIdx.x * 64;
  int n0 = blockIdx.y * 64;
  int mt  = wid >> 1;          // 0..3
  int ntp = (wid & 1) * 2;     // this wave owns nt = ntp, ntp+1
  v8f acc0 = {}; v8f acc1 = {};
  int nk = (K + 31) >> 5;

  int idx8 = tid * 8;          // this thread stages 8 contiguous elements per tile
  int row  = idx8 >> 5;
  int kk0  = idx8 & 31;
  int m    = m0 + row;
  const float* xrow = X + (size_t)(m >> 6) * xs_t + (size_t)(m & 63) * xs_b;
  const float* wrow = W + (size_t)(n0 + row) * ldw;

  for (int kb = 0; kb < nk; ++kb) {
    int k0 = kb * 32;
    if (k0 + 32 <= K) {               // uniform full-tile fast path
      const float* xp = xrow + k0 + kk0;
      const float* wp = wrow + k0 + kk0;
      float4 x0 = *(const float4*)xp, x1 = *(const float4*)(xp + 4);
      float4 w0 = *(const float4*)wp, w1 = *(const float4*)(wp + 4);
      v8bf xv, wv;
      xv[0]=(bf16_t)x0.x; xv[1]=(bf16_t)x0.y; xv[2]=(bf16_t)x0.z; xv[3]=(bf16_t)x0.w;
      xv[4]=(bf16_t)x1.x; xv[5]=(bf16_t)x1.y; xv[6]=(bf16_t)x1.z; xv[7]=(bf16_t)x1.w;
      wv[0]=(bf16_t)w0.x; wv[1]=(bf16_t)w0.y; wv[2]=(bf16_t)w0.z; wv[3]=(bf16_t)w0.w;
      wv[4]=(bf16_t)w1.x; wv[5]=(bf16_t)w1.y; wv[6]=(bf16_t)w1.z; wv[7]=(bf16_t)w1.w;
      *(v8bf*)&xs[idx8] = xv;
      *(v8bf*)&wt[idx8] = wv;
    } else {                           // tail (only K=40 input projections)
#pragma unroll
      for (int e = 0; e < 8; ++e) {
        int kk = kk0 + e;
        float xv = 0.f, wv = 0.f;
        if (k0 + kk < K) { xv = xrow[k0 + kk]; wv = wrow[k0 + kk]; }
        xs[idx8 + e] = (bf16_t)xv;
        wt[idx8 + e] = (bf16_t)wv;
      }
    }
    __syncthreads();
    v16bf a  = load_frag16(xs, mt * 16, 32, 0);
    v16bf b0 = load_frag16(wt, ntp * 16, 32, 0);
    v16bf b1 = load_frag16(wt, (ntp + 1) * 16, 32, 0);
    acc0 = WMMA_BF16(a, b0, acc0);
    acc1 = WMMA_BF16(a, b1, acc1);
    __syncthreads();
  }
  int nlo = lane & 15, hi8 = (lane >> 4) * 8;
#pragma unroll
  for (int r = 0; r < 8; ++r) {
    int mm = m0 + mt * 16 + r + hi8;
    int n  = n0 + ntp * 16 + nlo;
    float bs = (bias1 ? bias1[n] : 0.f) + (bias2 ? bias2[n] : 0.f);
    Out[(size_t)mm * N + n] = acc0[r] + bs;
    n += 16;
    bs = (bias1 ? bias1[n] : 0.f) + (bias2 ? bias2[n] : 0.f);
    Out[(size_t)mm * N + n] = acc1[r] + bs;
  }
}

// ---------------------------------------------------------------------------
// Persistent LSTM recurrence. Xproj = x@wih^T + bih + bhh precomputed.
// One 1024-thread block per direction; whh (pre-converted bf16) copied to LDS
// via raw uint4; h double-buffered bf16 in LDS. Wave (mt,ng) owns the 16x16
// (batch,unit) tile for all 4 gates -> lane-local gate fusion; c in registers.
// ---------------------------------------------------------------------------
__global__ __launch_bounds__(1024)
void k_lstm_rec(const float* xp0, const bf16_t* whh0, float* ho0, int dir0,
                const float* xp1, const bf16_t* whh1, float* ho1, int dir1, int T)
{
  extern __shared__ char smem[];
  bf16_t* whh_l = (bf16_t*)smem;                      // 512*128 bf16 = 128KB
  bf16_t* hbuf  = (bf16_t*)(smem + 512 * 128 * 2);    // 2 * 64*128 bf16 = 32KB
  const float*  xp  = (blockIdx.x == 0) ? xp0  : xp1;
  const bf16_t* whh = (blockIdx.x == 0) ? whh0 : whh1;
  float* hout       = (blockIdx.x == 0) ? ho0  : ho1;
  int dir           = (blockIdx.x == 0) ? dir0 : dir1;

  int tid = threadIdx.x, lane = tid & 31, wid = tid >> 5;
  int mt = wid >> 3, ng = wid & 7;
  int nlo = lane & 15, hi8 = (lane >> 4) * 8;

  {
    const uint4* src = (const uint4*)whh;
    uint4* dst = (uint4*)whh_l;
    for (int i = tid; i < 512 * 128 * 2 / 16; i += 1024) dst[i] = src[i];
    uint4 z = {0, 0, 0, 0};
    uint4* hz = (uint4*)hbuf;
    for (int i = tid; i < 2 * 64 * 128 * 2 / 16; i += 1024) hz[i] = z;
  }
  __syncthreads();

  float cst[8];
#pragma unroll
  for (int r = 0; r < 8; ++r) cst[r] = 0.f;

  int p = 0;
  for (int it = 0; it < T; ++it) {
    int t = dir ? (T - 1 - it) : it;
    const float* xpt = xp + (size_t)t * 64 * 512;
    bf16_t* hc = hbuf + p * 64 * 128;
    bf16_t* hn = hbuf + (p ^ 1) * 64 * 128;

    v16bf afr[4];
#pragma unroll
    for (int kc = 0; kc < 4; ++kc) afr[kc] = load_frag16(hc, mt * 16, 128, kc * 32);

    v8f acc[4];
#pragma unroll
    for (int gi = 0; gi < 4; ++gi) {
      int j0 = gi * 128 + ng * 16;
      v8f a;
#pragma unroll
      for (int r = 0; r < 8; ++r) {
        int b = mt * 16 + r + hi8;
        a[r] = xpt[(size_t)b * 512 + j0 + nlo];
      }
#pragma unroll
      for (int kc = 0; kc < 4; ++kc) {
        v16bf bf = load_frag16(whh_l, j0, 128, kc * 32);
        a = WMMA_BF16(afr[kc], bf, a);
      }
      acc[gi] = a;
    }
#pragma unroll
    for (int r = 0; r < 8; ++r) {
      int b = mt * 16 + r + hi8;
      int u = ng * 16 + nlo;
      float iv = sigf(acc[0][r]), fv = sigf(acc[1][r]);
      float gv = tanhf(acc[2][r]), ov = sigf(acc[3][r]);
      cst[r] = fv * cst[r] + iv * gv;
      float hv = ov * tanhf(cst[r]);
      hn[b * 128 + u] = (bf16_t)hv;
      hout[((size_t)t * 64 + b) * 128 + u] = hv;
    }
    __syncthreads();
    p ^= 1;
  }
}

// ---------------------------------------------------------------------------
// Fused attention decoder: 255 steps, one 1024-thread persistent block.
// Gates via WMMA over two recurrent inputs (context c, hidden sh), K=128 each,
// against bf16-packed wihd[:,E:] / whhd (L2-resident). hv pre-transposed to
// [b,d,tt] so the context reduction streams contiguously.
// ---------------------------------------------------------------------------
__global__ __launch_bounds__(1024)
void k_decoder(const float* __restrict__ yproj,   // [255,64,512] (incl. biases)
               const bf16_t* __restrict__ wcd,    // [512,128] = wihd[:,512:640]
               const bf16_t* __restrict__ whhd,   // [512,128]
               const float* __restrict__ hk,      // [(tt,b),128]
               const float* __restrict__ hvt,     // [(b,d),tt] transposed values
               const float* __restrict__ sh0, const float* __restrict__ sc0,
               const float* __restrict__ w1, const float* __restrict__ b1,
               const float* __restrict__ w2, const float* __restrict__ b2,
               const float* __restrict__ w3, const float* __restrict__ b3,
               float* __restrict__ out)           // [64,255,34]
{
  extern __shared__ char smem[];
  bf16_t* cbf  = (bf16_t*)smem;                 // 64*128 bf16
  bf16_t* shbf = cbf + 64 * 128;                // 64*128 bf16
  float*  scn  = (float*)(smem + 2 * 64 * 128 * 2);
  float*  shn  = scn + 64 * 128;
  float*  cnf  = shn + 64 * 128;
  float*  lgt  = cnf + 64 * 128;                // 64*512
  float*  rmax = lgt + 64 * 512;                // 64
  float*  rsum = rmax + 64;                     // 64
  float*  o1   = rsum + 64;                     // 64*34

  int tid = threadIdx.x, lane = tid & 31, wid = tid >> 5;
  int mt = wid >> 3, ng = wid & 7;
  int nlo = lane & 15, hi8 = (lane >> 4) * 8;

  for (int i = tid; i < 64 * 128; i += 1024) { cbf[i] = (bf16_t)0.f; shbf[i] = (bf16_t)sh0[i]; }
  float sreg[8];
#pragma unroll
  for (int r = 0; r < 8; ++r) {
    int b = mt * 16 + r + hi8, u = ng * 16 + nlo;
    sreg[r] = sc0[b * 128 + u];
  }
  __syncthreads();

  for (int t = 0; t < 255; ++t) {
    const float* ypt = yproj + (size_t)t * 64 * 512;
    v16bf ac[4], as[4];
#pragma unroll
    for (int kc = 0; kc < 4; ++kc) {
      ac[kc] = load_frag16(cbf,  mt * 16, 128, kc * 32);
      as[kc] = load_frag16(shbf, mt * 16, 128, kc * 32);
    }
    v8f acc[4];
#pragma unroll
    for (int gi = 0; gi < 4; ++gi) {
      int j0 = gi * 128 + ng * 16;
      v8f a;
#pragma unroll
      for (int r = 0; r < 8; ++r) {
        int b = mt * 16 + r + hi8;
        a[r] = ypt[(size_t)b * 512 + j0 + nlo];
      }
#pragma unroll
      for (int kc = 0; kc < 4; ++kc) {
        v16bf bc = load_frag16(wcd,  j0, 128, kc * 32);
        a = WMMA_BF16(ac[kc], bc, a);
        v16bf bs = load_frag16(whhd, j0, 128, kc * 32);
        a = WMMA_BF16(as[kc], bs, a);
      }
      acc[gi] = a;
    }
#pragma unroll
    for (int r = 0; r < 8; ++r) {
      int b = mt * 16 + r + hi8, u = ng * 16 + nlo;
      float iv = sigf(acc[0][r]), fv = sigf(acc[1][r]);
      float gv = tanhf(acc[2][r]), ov = sigf(acc[3][r]);
      sreg[r] = fv * sreg[r] + iv * gv;
      float sh_ = ov * tanhf(sreg[r]);
      scn[b * 128 + u]  = sreg[r];
      shn[b * 128 + u]  = sh_;
      shbf[b * 128 + u] = (bf16_t)sh_;
    }
    __syncthreads();

    // attention logits: lgt[b,tt] = <hk[tt,b,:], scn[b,:]> (attends with cell state, per ref)
    for (int idx = tid; idx < 64 * 512; idx += 1024) {
      int b = idx >> 9, tt = idx & 511;
      const float4* hk4 = (const float4*)(hk + ((size_t)tt * 64 + b) * 128);
      const float4* sp4 = (const float4*)(scn + b * 128);
      __builtin_prefetch(hk4, 0, 1);
      float s = 0.f;
#pragma unroll 8
      for (int d = 0; d < 32; ++d) {
        float4 h4 = hk4[d], s4 = sp4[d];
        s += h4.x * s4.x + h4.y * s4.y + h4.z * s4.z + h4.w * s4.w;
      }
      lgt[b * 512 + tt] = s;
    }
    __syncthreads();

    // softmax reductions: each wave handles 2 batches, 16 lanes per batch
    {
      int b = wid * 2 + (lane >> 4);
      const float* lp = lgt + b * 512;
      float mx = -3.0e38f;
      for (int tt = nlo; tt < 512; tt += 16) mx = fmaxf(mx, lp[tt]);
#pragma unroll
      for (int mm = 1; mm < 16; mm <<= 1) mx = fmaxf(mx, __shfl_xor(mx, mm, 32));
      float sm = 0.f;
      for (int tt = nlo; tt < 512; tt += 16) sm += __expf(lp[tt] - mx);
#pragma unroll
      for (int mm = 1; mm < 16; mm <<= 1) sm += __shfl_xor(sm, mm, 32);
      if (nlo == 0) { rmax[b] = mx; rsum[b] = sm; }
    }
    __syncthreads();
    for (int idx = tid; idx < 64 * 512; idx += 1024) {
      int b = idx >> 9;
      lgt[idx] = __expf(lgt[idx] - rmax[b]) / rsum[b];
    }
    __syncthreads();

    // context c_n[b,d] = sum_tt att[b,tt] * hvt[(b,d),tt]   (contiguous stream)
    for (int idx = tid; idx < 64 * 128; idx += 1024) {
      int b = idx >> 7;
      const float4* ap = (const float4*)(lgt + b * 512);
      const float4* vp = (const float4*)(hvt + (size_t)idx * 512);
      float s = 0.f;
#pragma unroll 8
      for (int q = 0; q < 128; ++q) {
        float4 a4 = ap[q], v4 = vp[q];
        s += a4.x * v4.x + a4.y * v4.y + a4.z * v4.z + a4.w * v4.w;
      }
      cnf[idx] = s;
      cbf[idx] = (bf16_t)s;
    }
    __syncthreads();

    // out1 = relu(sh_n @ w1^T + b1 + c_n @ w2^T + b2)
    for (int idx = tid; idx < 64 * 34; idx += 1024) {
      int b = idx / 34, v = idx % 34;
      const float4* sp = (const float4*)(shn + b * 128);
      const float4* cp = (const float4*)(cnf + b * 128);
      const float4* w1p = (const float4*)(w1 + v * 128);
      const float4* w2p = (const float4*)(w2 + v * 128);
      float s = b1[v] + b2[v];
#pragma unroll 4
      for (int q = 0; q < 32; ++q) {
        float4 a4 = sp[q], b4 = w1p[q], c4 = cp[q], d4 = w2p[q];
        s += a4.x * b4.x + a4.y * b4.y + a4.z * b4.z + a4.w * b4.w;
        s += c4.x * d4.x + c4.y * d4.y + c4.z * d4.z + c4.w * d4.w;
      }
      o1[idx] = fmaxf(s, 0.f);
    }
    __syncthreads();

    // preds = softmax(out1 @ w3^T + b3)
    if (tid < 64) {
      int b = tid;
      float l2[34]; float mx = -3.0e38f;
      for (int v = 0; v < 34; ++v) {
        const float* w3p = w3 + v * 34; const float* op = o1 + b * 34;
        float s = b3[v];
        for (int u = 0; u < 34; ++u) s += op[u] * w3p[u];
        l2[v] = s; mx = fmaxf(mx, s);
      }
      float sm = 0.f;
      for (int v = 0; v < 34; ++v) { l2[v] = __expf(l2[v] - mx); sm += l2[v]; }
      float inv = 1.f / sm;
      float* op = out + ((size_t)b * 255 + t) * 34;
      for (int v = 0; v < 34; ++v) op[v] = l2[v] * inv;
    }
    __syncthreads();
  }
}

// ---------------------------------------------------------------------------
// Repack / gather helpers (4 elements per thread, float4 where contiguous)
// ---------------------------------------------------------------------------
__global__ void k_pack_conv1_x(const float* __restrict__ Hf, const float* __restrict__ Hr,
                               float* __restrict__ Xc, int total4) {
  int i4 = blockIdx.x * blockDim.x + threadIdx.x;
  if (i4 >= total4) return;
  int idx = i4 * 4;
  int kk = idx & 511, m = idx >> 9;
  int b = m & 63, tp = m >> 6;
  int k = kk >> 8, ci = kk & 255;
  int ts = 2 * tp + k;
  const float* src = (ci < 128) ? (Hf + ((size_t)ts * 64 + b) * 128 + ci)
                                : (Hr + ((size_t)ts * 64 + b) * 128 + (ci - 128));
  *(float4*)(Xc + idx) = *(const float4*)src;
}
__global__ void k_pack_conv2_x(const float* __restrict__ H1, float* __restrict__ Xc, int total4) {
  int i4 = blockIdx.x * blockDim.x + threadIdx.x;
  if (i4 >= total4) return;
  int idx = i4 * 4;
  int kk = idx & 255, m = idx >> 8;
  int b = m & 63, tp = m >> 6;
  int k = kk >> 7, ci = kk & 127;
  *(float4*)(Xc + idx) = *(const float4*)(H1 + ((size_t)(2 * tp + k) * 64 + b) * 128 + ci);
}
__global__ void k_pack_w1(const float* __restrict__ c1w, float* __restrict__ Wc, int total) {
  int idx = blockIdx.x * blockDim.x + threadIdx.x;
  if (idx >= total) return;
  int kk = idx & 511, co = idx >> 9;
  int k = kk >> 8, ci = kk & 255;
  Wc[idx] = c1w[co * 512 + ci * 2 + k];
}
__global__ void k_pack_w2(const float* __restrict__ c2w, float* __restrict__ Wc, int total) {
  int idx = blockIdx.x * blockDim.x + threadIdx.x;
  if (idx >= total) return;
  int kk = idx & 255, co = idx >> 8;
  int k = kk >> 7, ci = kk & 127;
  Wc[idx] = c2w[co * 256 + ci * 2 + k];
}
__global__ void k_gather_emb(const int* __restrict__ targets, const float* __restrict__ emb,
                             float* __restrict__ Y, int total4) {
  int i4 = blockIdx.x * blockDim.x + threadIdx.x;
  if (i4 >= total4) return;
  int idx = i4 * 4;
  int e = idx & 511, m = idx >> 9;
  int b = m & 63, t = m >> 6;
  int tok = targets[b * 256 + t];
  *(float4*)(Y + idx) = *(const float4*)(emb + (size_t)tok * 512 + e);
}
__global__ void k_cvt_bf16(const float* __restrict__ in, bf16_t* __restrict__ o, int n) {
  int idx = blockIdx.x * blockDim.x + threadIdx.x;
  if (idx < n) o[idx] = (bf16_t)in[idx];
}
__global__ void k_pack_wcd(const float* __restrict__ wihd, bf16_t* __restrict__ o, int total) {
  int idx = blockIdx.x * blockDim.x + threadIdx.x;
  if (idx >= total) return;
  int k = idx & 127, n = idx >> 7;
  o[idx] = (bf16_t)wihd[n * 640 + 512 + k];
}
__global__ void k_transpose_hv(const float* __restrict__ HV, float* __restrict__ HVT, int total) {
  int idx = blockIdx.x * blockDim.x + threadIdx.x;  // over (b*128+d)*512 + tt
  if (idx >= total) return;
  int tt = idx & 511, r = idx >> 9;
  int b = r >> 7, d = r & 127;
  HVT[idx] = HV[((size_t)tt * 64 + b) * 128 + d];
}

// ---------------------------------------------------------------------------
// Host orchestration
// ---------------------------------------------------------------------------
extern "C" void kernel_launch(void* const* d_in, const int* in_sizes, int n_in,
                              void* d_out, int out_size, void* d_ws, size_t ws_size,
                              hipStream_t stream) {
  (void)in_sizes; (void)n_in; (void)out_size; (void)ws_size;
  const float* utter  = (const float*)d_in[0];
  const int*   targets= (const int*)  d_in[1];
  const float* wih_f  = (const float*)d_in[2];  const float* whh_f = (const float*)d_in[3];
  const float* bih_f  = (const float*)d_in[4];  const float* bhh_f = (const float*)d_in[5];
  const float* wih_r  = (const float*)d_in[6];  const float* whh_r = (const float*)d_in[7];
  const float* bih_r  = (const float*)d_in[8];  const float* bhh_r = (const float*)d_in[9];
  const float* c1w    = (const float*)d_in[10]; const float* c1b   = (const float*)d_in[11];
  const float* wih1   = (const float*)d_in[12]; const float* whh1  = (const float*)d_in[13];
  const float* bih1   = (const float*)d_in[14]; const float* bhh1  = (const float*)d_in[15];
  const float* c2w    = (const float*)d_in[16]; const float* c2b   = (const float*)d_in[17];
  const float* wih2   = (const float*)d_in[18]; const float* whh2  = (const float*)d_in[19];
  const float* bih2   = (const float*)d_in[20]; const float* bhh2  = (const float*)d_in[21];
  const float* kw     = (const float*)d_in[22]; const float* kb    = (const float*)d_in[23];
  const float* vw     = (const float*)d_in[24]; const float* vb    = (const float*)d_in[25];
  const float* wihd   = (const float*)d_in[26]; const float* whhd  = (const float*)d_in[27];
  const float* bihd   = (const float*)d_in[28]; const float* bhhd  = (const float*)d_in[29];
  const float* w1     = (const float*)d_in[30]; const float* b1    = (const float*)d_in[31];
  const float* w2     = (const float*)d_in[32]; const float* b2    = (const float*)d_in[33];
  const float* w3     = (const float*)d_in[34]; const float* b3    = (const float*)d_in[35];
  const float* emb    = (const float*)d_in[36];
  const float* sh0    = (const float*)d_in[37]; const float* sc0   = (const float*)d_in[38];
  float* out = (float*)d_out;

  // workspace layout (big regions reused sequentially; lifetimes disjoint via stream order)
  const size_t SZ_XP = (size_t)2048 * 64 * 512 * 4;   // 256MB
  const size_t SZ_H  = (size_t)2048 * 64 * 128 * 4;   // 64MB
  char* w = (char*)d_ws;
  float* XF = (float*)(w);
  float* XR = (float*)(w + SZ_XP);
  float* HF = (float*)(w + 2 * SZ_XP);
  float* HR = (float*)(w + 2 * SZ_XP + SZ_H);
  char* smalls = w + 2 * SZ_XP + 2 * SZ_H;
  float*  WC1  = (float*)(smalls);                          // 128*512 f32
  float*  WC2  = (float*)(smalls + 262144);                 // 128*256 f32
  bf16_t* WHHD = (bf16_t*)(smalls + 393216);                // 512*128 bf16 each below
  bf16_t* WCD  = (bf16_t*)(smalls + 524288);
  bf16_t* WHHF = (bf16_t*)(smalls + 655360);
  bf16_t* WHHR = (bf16_t*)(smalls + 786432);
  bf16_t* WHH1 = (bf16_t*)(smalls + 917504);
  bf16_t* WHH2 = (bf16_t*)(smalls + 1048576);
  // aliases
  float* XC1 = XF;   float* C1O = XR;   float* XP1 = XF;  float* H1 = HF;
  float* XC2 = XR;   float* C2O = HR;   float* XP2 = XF;  float* H2 = HF;
  float* HK  = XR;
  float* HV  = (float*)((char*)XR + (size_t)512 * 64 * 128 * 4);
  float* HVT = (float*)((char*)XR + (size_t)2 * 512 * 64 * 128 * 4);
  float* YEMB = XF;  float* YPROJ = (float*)((char*)XF + ((size_t)64 << 20));

  const size_t REC_LDS = 512 * 128 * 2 + 2 * 64 * 128 * 2;                 // 160KB
  const size_t DEC_LDS = 2*64*128*2 + 3*64*128*4 + 64*512*4 + 2*64*4 + 64*34*4;
  hipFuncSetAttribute((const void*)k_lstm_rec, hipFuncAttributeMaxDynamicSharedMemorySize, (int)REC_LDS);
  hipFuncSetAttribute((const void*)k_decoder,  hipFuncAttributeMaxDynamicSharedMemorySize, (int)DEC_LDS);

  dim3 blk(256);
  const int NW = 512 * 128;
  // recurrent weights -> bf16 (one-time)
  k_cvt_bf16<<<(NW + 255) / 256, blk, 0, stream>>>(whh_f, WHHF, NW);
  k_cvt_bf16<<<(NW + 255) / 256, blk, 0, stream>>>(whh_r, WHHR, NW);
  k_cvt_bf16<<<(NW + 255) / 256, blk, 0, stream>>>(whh1,  WHH1, NW);
  k_cvt_bf16<<<(NW + 255) / 256, blk, 0, stream>>>(whh2,  WHH2, NW);
  k_cvt_bf16<<<(NW + 255) / 256, blk, 0, stream>>>(whhd,  WHHD, NW);
  k_pack_wcd<<<(NW + 255) / 256, blk, 0, stream>>>(wihd, WCD, NW);
  // 1-2. BiLSTM input projections, all timesteps at once (utter [B,T,40] via gather strides)
  k_gemm<<<dim3(2048, 8), blk, 0, stream>>>(utter, 40L, 2048L * 40L, wih_f, 40, bih_f, bhh_f, XF, 512, 40);
  k_gemm<<<dim3(2048, 8), blk, 0, stream>>>(utter, 40L, 2048L * 40L, wih_r, 40, bih_r, bhh_r, XR, 512, 40);
  // 3. BiLSTM recurrence, both directions in parallel
  k_lstm_rec<<<2, 1024, REC_LDS, stream>>>(XF, WHHF, HF, 0, XR, WHHR, HR, 1, 2048);
  // 4-6. conv1 (stride-2, k=2) as GEMM via im2col pack
  { int n4 = 1024 * 64 * 512 / 4; k_pack_conv1_x<<<(n4 + 255) / 256, blk, 0, stream>>>(HF, HR, XC1, n4); }
  { int n = 128 * 512;            k_pack_w1<<<(n + 255) / 256, blk, 0, stream>>>(c1w, WC1, n); }
  k_gemm<<<dim3(1024, 2), blk, 0, stream>>>(XC1, 64L * 512L, 512L, WC1, 512, c1b, nullptr, C1O, 128, 512);
  // 7-8. LSTM layer 1
  k_gemm<<<dim3(1024, 8), blk, 0, stream>>>(C1O, 64L * 128L, 128L, wih1, 128, bih1, bhh1, XP1, 512, 128);
  k_lstm_rec<<<1, 1024, REC_LDS, stream>>>(XP1, WHH1, H1, 0, XP1, WHH1, H1, 0, 1024);
  // 9-11. conv2 as GEMM
  { int n4 = 512 * 64 * 256 / 4; k_pack_conv2_x<<<(n4 + 255) / 256, blk, 0, stream>>>(H1, XC2, n4); }
  { int n = 128 * 256;           k_pack_w2<<<(n + 255) / 256, blk, 0, stream>>>(c2w, WC2, n); }
  k_gemm<<<dim3(512, 2), blk, 0, stream>>>(XC2, 64L * 256L, 256L, WC2, 256, c2b, nullptr, C2O, 128, 256);
  // 12-13. LSTM layer 2
  k_gemm<<<dim3(512, 8), blk, 0, stream>>>(C2O, 64L * 128L, 128L, wih2, 128, bih2, bhh2, XP2, 512, 128);
  k_lstm_rec<<<1, 1024, REC_LDS, stream>>>(XP2, WHH2, H2, 0, XP2, WHH2, H2, 0, 512);
  // 14-15. keys/values (+ transpose values for contiguous context reduction)
  k_gemm<<<dim3(512, 2), blk, 0, stream>>>(H2, 64L * 128L, 128L, kw, 128, kb, nullptr, HK, 128, 128);
  k_gemm<<<dim3(512, 2), blk, 0, stream>>>(H2, 64L * 128L, 128L, vw, 128, vb, nullptr, HV, 128, 128);
  { int n = 64 * 128 * 512; k_transpose_hv<<<(n + 255) / 256, blk, 0, stream>>>(HV, HVT, n); }
  // 16-17. decoder input projection (embedding gather + GEMM over wihd[:, :512], biases folded)
  { int n4 = 255 * 64 * 512 / 4; k_gather_emb<<<(n4 + 255) / 256, blk, 0, stream>>>(targets, emb, YEMB, n4); }
  k_gemm<<<dim3(255, 8), blk, 0, stream>>>(YEMB, 64L * 512L, 512L, wihd, 640, bihd, bhhd, YPROJ, 512, 512);
  // 18. fused attention decoder
  k_decoder<<<1, 1024, DEC_LDS, stream>>>(YPROJ, WCD, WHHD, HK, HVT, sh0, sc0,
                                          w1, b1, w2, b2, w3, b3, out);
}
